// MulHeadAttn_71657234366595
// MI455X (gfx1250) — matrix operand order repaired and checked by reading, not compiled, not run on hardware
//
#include <hip/hip_runtime.h>
#include <hip/hip_bf16.h>
#include <stdint.h>

// MI455X (gfx1250) fused multi-head attention, bf16 WMMA with f32 accumulate.
// (0) one-shot f32->bf16 conversion of x, W_qkv, W_o
// (1) QKV GEMM (LDS-free, direct-global WMMA fragments; Q,K row-major bf16; V transposed bf16)
// (2) flash attention (online softmax, all matmuls on WMMA)
// (3) output projection GEMM + bias (LDS-free)

#define EMBED 1024
#define NH    16
#define HD    64
#define SEQ   2048
#define BATCH 2

typedef __attribute__((ext_vector_type(16))) __bf16 v16bf;
typedef __attribute__((ext_vector_type(8)))  float  v8f;

// f32 -> bf16 round-to-nearest-even, integer domain.
__device__ __forceinline__ uint32_t bfround(float f) {
    union { float f; uint32_t u; } c; c.f = f;
    return c.u + 0x7FFFu + ((c.u >> 16) & 1u);
}
// pack two floats into {bf16(b) hi, bf16(a) lo} with one v_perm_b32
__device__ __forceinline__ uint32_t pkbf(float a, float b) {
    return __builtin_amdgcn_perm(bfround(b), bfround(a), 0x07060302u);
}
__device__ __forceinline__ uint4 pack8(float4 lo, float4 hi) {
    return make_uint4(pkbf(lo.x, lo.y), pkbf(lo.z, lo.w), pkbf(hi.x, hi.y), pkbf(hi.z, hi.w));
}
__device__ __forceinline__ uint16_t bf16u(float f) { return (uint16_t)(bfround(f) >> 16); }

// A-fragment (16x32 bf16, M x K) from row-major [rows][ldm] bf16.
// ISA 7.12.2: lanes 0-15 (M=lane) hold K=[0..7],[16..23]; lanes 16-31 hold K=[8..15],[24..31].
__device__ __forceinline__ v16bf load_afrag(const __bf16* base, int ldm, int row, int k0, int lane) {
    int hl = lane >> 4, lo = lane & 15;
    const __bf16* p = base + (size_t)(row + lo) * ldm + k0 + 8 * hl;
    union { v16bf v; uint4 u[2]; } r;
    r.u[0] = *(const uint4*)p;
    r.u[1] = *(const uint4*)(p + 16);
    return r.v;
}

// B-fragment (32x16 bf16, K x N): B[k][n] = base[n*ldn + k] (K contiguous per lane).
// ISA: lanes 0-15 (N=lane) hold K=[0..7],[16..23]; lanes 16-31 hold K=[8..15],[24..31].
__device__ __forceinline__ v16bf load_bfrag(const __bf16* base, int ldn, int col, int k0, int lane) {
    int hl = lane >> 4, lo = lane & 15;
    const __bf16* p = base + (size_t)(col + lo) * ldn + k0 + 8 * hl;
    union { v16bf v; uint4 u[2]; } r;
    r.u[0] = *(const uint4*)p;
    r.u[1] = *(const uint4*)(p + 16);
    return r.v;
}

__device__ __forceinline__ v8f wmma_bf16(v16bf a, v16bf b, v8f c) {
    return __builtin_amdgcn_wmma_f32_16x16x32_bf16(false, a, false, b, (short)0, c, false, false);
}

// ---------------------------------------------------------------------------
// Kernel 0: elementwise f32 -> bf16 (8 elements / thread)
// ---------------------------------------------------------------------------
__global__ __launch_bounds__(256) void to_bf16(const float* __restrict__ in,
                                               uint4* __restrict__ out) {
    const size_t i = (size_t)blockIdx.x * 256 + threadIdx.x;
    const float4* p = (const float4*)(in + i * 8);
    out[i] = pack8(p[0], p[1]);
}

// ---------------------------------------------------------------------------
// Kernel 1: qkv = x @ W_qkv^T   (M=4096, N=3072, K=1024), all-bf16 inputs.
// WG 256 thr (8 waves), tile 128x128, wave tile 32x64 (2x4 WMMA accs).
// LDS-free: fragments loaded directly from global in native WMMA layout.
// Writes: Q,K bf16 [b][h][c][64]; V transposed bf16 [b][h][64][c].
// ---------------------------------------------------------------------------
__global__ __launch_bounds__(256) void qkv_gemm(const __bf16* __restrict__ xb,
                                                const __bf16* __restrict__ Wb,
                                                __bf16* __restrict__ Qb,
                                                __bf16* __restrict__ Kb,
                                                __bf16* __restrict__ Vt) {
    const int tid  = threadIdx.x;
    const int lane = tid & 31, wid = tid >> 5;
    const int mw = (wid >> 1) * 32, nw = (wid & 1) * 64;
    const int m0 = blockIdx.y * 128, n0 = blockIdx.x * 128;
    const int hl = lane >> 4, lo = lane & 15;

    v8f acc[2][4] = {};

    for (int kt = 0; kt < EMBED; kt += 32) {
        v16bf af[2], bf[4];
        #pragma unroll
        for (int i = 0; i < 2; i++) af[i] = load_afrag(xb, EMBED, m0 + mw + i * 16, kt, lane);
        #pragma unroll
        for (int j = 0; j < 4; j++) bf[j] = load_bfrag(Wb, EMBED, n0 + nw + j * 16, kt, lane);
        #pragma unroll
        for (int i = 0; i < 2; i++)
            #pragma unroll
            for (int j = 0; j < 4; j++)
                acc[i][j] = wmma_bf16(af[i], bf[j], acc[i][j]);
    }

    // Epilogue: route Q/K/V thirds to their buffers (bf16).
    #pragma unroll
    for (int i = 0; i < 2; i++) {
        #pragma unroll
        for (int j = 0; j < 4; j++) {
            const int n     = n0 + nw + j * 16 + lo;
            const int mbase = m0 + mw + i * 16 + 8 * hl;   // rows mbase+v, v=0..7
            if (n < 2 * EMBED) {
                uint16_t* dst = (uint16_t*)((n < EMBED) ? Qb : Kb);
                const int f = n & (EMBED - 1);
                const int h = f >> 6, d = f & 63;
                #pragma unroll
                for (int v = 0; v < 8; v++) {
                    const int m = mbase + v;
                    const int b = m >> 11, c = m & (SEQ - 1);
                    dst[(((size_t)(b * NH + h) * SEQ + c) << 6) + d] = bf16u(acc[i][j][v]);
                }
            } else {
                const int f = n - 2 * EMBED;
                const int h = f >> 6, d = f & 63;
                const int b = mbase >> 11, c = mbase & (SEQ - 1);
                uint4 pk;
                pk.x = pkbf(acc[i][j][0], acc[i][j][1]);
                pk.y = pkbf(acc[i][j][2], acc[i][j][3]);
                pk.z = pkbf(acc[i][j][4], acc[i][j][5]);
                pk.w = pkbf(acc[i][j][6], acc[i][j][7]);
                *(uint4*)&Vt[((size_t)(b * NH + h) * HD + d) * SEQ + c] = pk;  // 8 consecutive c
            }
        }
    }
}

// ---------------------------------------------------------------------------
// Kernel 2: flash attention. Block = 4 waves (128 thr) = one (b,h) x 64 q-rows.
// Each wave owns a 16-row q tile; streams 32 keys/iter; all matmuls on WMMA.
// ---------------------------------------------------------------------------
__global__ __launch_bounds__(128) void flash_attn(const __bf16* __restrict__ Qb,
                                                  const __bf16* __restrict__ Kb,
                                                  const __bf16* __restrict__ Vt,
                                                  __bf16* __restrict__ Ao) {
    __shared__ uint16_t Ps[4][16][32];               // per-wave P tile (16 q x 32 keys)

    const int tid  = threadIdx.x;
    const int lane = tid & 31, w = tid >> 5;
    const int hl   = lane >> 4, lo = lane & 15;

    const int bidx = blockIdx.x;
    const int qb = bidx & 31;
    const int h  = (bidx >> 5) & (NH - 1);
    const int b  = bidx >> 9;
    const int bh = b * NH + h;

    const __bf16* Q = Qb + (size_t)bh * SEQ * HD;
    const __bf16* K = Kb + (size_t)bh * SEQ * HD;
    const __bf16* V = Vt + (size_t)bh * HD * SEQ;    // transposed: [d][c]
    const int q0 = qb * 64 + w * 16;

    v16bf qf[2];
    qf[0] = load_afrag(Q, HD, q0, 0, lane);
    qf[1] = load_afrag(Q, HD, q0, 32, lane);

    v8f o[4] = {};
    float mrun[8], lrun[8];
    #pragma unroll
    for (int v = 0; v < 8; v++) { mrun[v] = -1e30f; lrun[v] = 0.0f; }

    for (int kb = 0; kb < SEQ; kb += 32) {
        // scores: two 16x16 tiles (keys kb..+15, kb+16..+31), K-dim = 64 (2 WMMAs each)
        v8f s0 = {}, s1 = {};
        s0 = wmma_bf16(qf[0], load_bfrag(K, HD, kb,      0,  lane), s0);
        s0 = wmma_bf16(qf[1], load_bfrag(K, HD, kb,      32, lane), s0);
        s1 = wmma_bf16(qf[0], load_bfrag(K, HD, kb + 16, 0,  lane), s1);
        s1 = wmma_bf16(qf[1], load_bfrag(K, HD, kb + 16, 32, lane), s1);

        // online softmax update per q-row (row = v + 8*hl, spread over 16 N-lanes)
        #pragma unroll
        for (int v = 0; v < 8; v++) {
            float a  = s0[v] * 0.125f;               // 1/sqrt(64)
            float bb = s1[v] * 0.125f;
            float rmax = fmaxf(a, bb);
            #pragma unroll
            for (int off = 8; off > 0; off >>= 1) rmax = fmaxf(rmax, __shfl_xor(rmax, off, 32));
            const float mnew  = fmaxf(mrun[v], rmax);
            const float alpha = __expf(mrun[v] - mnew);
            const float p0 = __expf(a - mnew);
            const float p1 = __expf(bb - mnew);
            float rs = p0 + p1;
            #pragma unroll
            for (int off = 8; off > 0; off >>= 1) rs += __shfl_xor(rs, off, 32);
            lrun[v] = lrun[v] * alpha + rs;
            mrun[v] = mnew;
            #pragma unroll
            for (int j = 0; j < 4; j++) o[j][v] *= alpha;
            Ps[w][v + 8 * hl][lo]      = bf16u(p0);  // re-layout acc -> A-frag via LDS
            Ps[w][v + 8 * hl][16 + lo] = bf16u(p1);
        }
        __syncthreads();

        const v16bf pf = load_afrag((const __bf16*)&Ps[w][0][0], 32, 0, 0, lane);
        #pragma unroll
        for (int j = 0; j < 4; j++) {
            // B-frag from V^T: lane n = output channel d, keys kb+8*hl+[0..7] and kb+16+8*hl+[0..7]
            o[j] = wmma_bf16(pf, load_bfrag(V, SEQ, j * 16, kb, lane), o[j]);
        }
        __syncthreads();
    }

    // normalize and scatter to attn-out [b][c][E] bf16
    #pragma unroll
    for (int j = 0; j < 4; j++) {
        const int d = h * 64 + j * 16 + lo;
        #pragma unroll
        for (int v = 0; v < 8; v++) {
            const int c = q0 + v + 8 * hl;
            ((uint16_t*)Ao)[((size_t)(b * SEQ + c)) * EMBED + d] = bf16u(o[j][v] / lrun[v]);
        }
    }
}

// ---------------------------------------------------------------------------
// Kernel 3: out = attn @ W_o^T + b_o   (M=4096, N=1024, K=1024), f32 output.
// LDS-free, direct-global WMMA fragments (both operands bf16).
// ---------------------------------------------------------------------------
__global__ __launch_bounds__(256) void out_gemm(const __bf16* __restrict__ A,
                                                const __bf16* __restrict__ Wb,
                                                const float* __restrict__ bo,
                                                float* __restrict__ out) {
    const int tid  = threadIdx.x;
    const int lane = tid & 31, wid = tid >> 5;
    const int mw = (wid >> 1) * 32, nw = (wid & 1) * 64;
    const int m0 = blockIdx.y * 128, n0 = blockIdx.x * 128;
    const int hl = lane >> 4, lo = lane & 15;

    v8f acc[2][4] = {};

    for (int kt = 0; kt < EMBED; kt += 32) {
        v16bf af[2], bf[4];
        #pragma unroll
        for (int i = 0; i < 2; i++) af[i] = load_afrag(A, EMBED, m0 + mw + i * 16, kt, lane);
        #pragma unroll
        for (int j = 0; j < 4; j++) bf[j] = load_bfrag(Wb, EMBED, n0 + nw + j * 16, kt, lane);
        #pragma unroll
        for (int i = 0; i < 2; i++)
            #pragma unroll
            for (int j = 0; j < 4; j++)
                acc[i][j] = wmma_bf16(af[i], bf[j], acc[i][j]);
    }

    #pragma unroll
    for (int i = 0; i < 2; i++) {
        #pragma unroll
        for (int j = 0; j < 4; j++) {
            const int n = n0 + nw + j * 16 + lo;
            const float bias = bo[n];
            const int mbase = m0 + mw + i * 16 + 8 * hl;
            #pragma unroll
            for (int v = 0; v < 8; v++)
                out[(size_t)(mbase + v) * EMBED + n] = acc[i][j][v] + bias;
        }
    }
}

// ---------------------------------------------------------------------------
extern "C" void kernel_launch(void* const* d_in, const int* in_sizes, int n_in,
                              void* d_out, int out_size, void* d_ws, size_t ws_size,
                              hipStream_t stream) {
    (void)in_sizes; (void)n_in; (void)out_size; (void)ws_size;
    const float* x    = (const float*)d_in[0];   // [2,2048,1024]
    const float* Wqkv = (const float*)d_in[1];   // [3072,1024]
    const float* Wo   = (const float*)d_in[2];   // [1024,1024]
    const float* bo   = (const float*)d_in[3];   // [1024]
    float* out = (float*)d_out;                  // [2,2048,1024] f32

    // Workspace (bf16 elements):
    //   Q | K | V^T | attn-out (4 x 4,194,304) then x_bf | Wqkv_bf | Wo_bf
    const size_t QK_ELEMS = (size_t)BATCH * NH * SEQ * HD;           // 4,194,304
    const size_t NX   = (size_t)BATCH * SEQ * EMBED;                 // 4,194,304
    const size_t NWQ  = (size_t)3 * EMBED * EMBED;                   // 3,145,728
    const size_t NWO  = (size_t)EMBED * EMBED;                       // 1,048,576
    __bf16* Qb  = (__bf16*)d_ws;
    __bf16* Kb  = Qb + QK_ELEMS;
    __bf16* Vt  = Kb + QK_ELEMS;
    __bf16* Ao  = Vt + QK_ELEMS;
    __bf16* xb  = Ao + QK_ELEMS;
    __bf16* Wqb = xb + NX;
    __bf16* Wob = Wqb + NWQ;

    to_bf16<<<dim3(NX  / (8 * 256)), 256, 0, stream>>>(x,    (uint4*)xb);
    to_bf16<<<dim3(NWQ / (8 * 256)), 256, 0, stream>>>(Wqkv, (uint4*)Wqb);
    to_bf16<<<dim3(NWO / (8 * 256)), 256, 0, stream>>>(Wo,   (uint4*)Wob);

    qkv_gemm<<<dim3(3 * EMBED / 128, (BATCH * SEQ) / 128), 256, 0, stream>>>(xb, Wqb, Qb, Kb, Vt);
    flash_attn<<<dim3(BATCH * NH * (SEQ / 64)), 128, 0, stream>>>(Qb, Kb, Vt, Ao);
    out_gemm<<<dim3(EMBED / 128, (BATCH * SEQ) / 128), 256, 0, stream>>>(Ao, Wob, bo, out);
}
